// CenterLoss_85985245266075
// MI455X (gfx1250) — matrix operand order repaired and checked by reading, not compiled, not run on hardware
//
#include <hip/hip_runtime.h>
#include <hip/hip_bf16.h>

typedef __attribute__((ext_vector_type(16))) __bf16 v16bf;
typedef __attribute__((ext_vector_type(8)))  __bf16 v8bf;
typedef __attribute__((ext_vector_type(4)))  __bf16 v4bf;
typedef __attribute__((ext_vector_type(8)))  float  v8f;

#define BATCH 65536
#define FEAT  1024
#define NCLS  96
#define CT_TILES 6            // 96 / 16
#define WAVES_PER_BLOCK 8
#define ROWS_PER_WAVE  16
#define ROWS_PER_BLOCK (WAVES_PER_BLOCK * ROWS_PER_WAVE)   // 128
#define NBLOCKS (BATCH / ROWS_PER_BLOCK)                    // 512

#define K_CHUNK   128                       // K-extent staged in LDS per buffer
#define N_CHUNKS  (FEAT / K_CHUNK)          // 8
#define LDS_ROW_B 272                       // 128 bf16 = 256B + 16B pad (bank-conflict-free)
#define CHUNK_B   (NCLS * LDS_ROW_B)        // 26112 B per buffer

// ---------------------------------------------------------------------------
// CDNA5 async copy helpers (ASYNCcnt path)
// ---------------------------------------------------------------------------
__device__ __forceinline__ void async_load_b128(const void* gsrc, unsigned lds_off) {
  // global_load_async_to_lds_b128 vdst(lds byte addr), vaddr(64b global), off
  asm volatile("global_load_async_to_lds_b128 %0, %1, off"
               :: "v"(lds_off), "v"((unsigned long long)(size_t)gsrc)
               : "memory");
}

__device__ __forceinline__ void async_wait0() {
#if __has_builtin(__builtin_amdgcn_s_wait_asynccnt)
  __builtin_amdgcn_s_wait_asynccnt(0);
#else
  asm volatile("s_wait_asynccnt 0" ::: "memory");
#endif
}

// ---------------------------------------------------------------------------
// Kernel 1: centers f32 -> bf16 (WMMA B operand) + ||c||^2 in f32
// ---------------------------------------------------------------------------
__global__ __launch_bounds__(256) void centerloss_prep(
    const float* __restrict__ centers,
    __bf16* __restrict__ ctr_bf,
    float* __restrict__ csq) {
  const int c = blockIdx.x;     // one block per class
  const int t = threadIdx.x;    // 256 threads * 4 floats = 1024
  const float4 v = ((const float4*)(centers + (size_t)c * FEAT))[t];
  v4bf o;
  o[0] = (__bf16)v.x; o[1] = (__bf16)v.y; o[2] = (__bf16)v.z; o[3] = (__bf16)v.w;
  ((v4bf*)(ctr_bf + (size_t)c * FEAT))[t] = o;

  float s = v.x * v.x + v.y * v.y + v.z * v.z + v.w * v.w;
#pragma unroll
  for (int off = 16; off > 0; off >>= 1) s += __shfl_xor(s, off, 32);
  __shared__ float red[8];
  if ((t & 31) == 0) red[t >> 5] = s;
  __syncthreads();
  if (t == 0) {
    float tot = 0.f;
#pragma unroll
    for (int w = 0; w < 8; ++w) tot += red[w];
    csq[c] = tot;
  }
}

// ---------------------------------------------------------------------------
// Kernel 2: fused WMMA GEMM (features x centers^T) + masked clamp-reduce.
// B (centers, bf16) is double-buffered in LDS via async-load-to-LDS DMA;
// one wave = 16 rows x 96 cols, K staged in chunks of 128.
// ---------------------------------------------------------------------------
__global__ __launch_bounds__(256) void centerloss_main(
    const float*  __restrict__ features,
    const int*    __restrict__ labels,
    const __bf16* __restrict__ ctr_bf,
    const float*  __restrict__ csq,
    float*        __restrict__ partials) {
  const int tid   = threadIdx.x;
  const int wid   = tid >> 5;
  const int lane  = tid & 31;
  const int row16 = lane & 15;   // A: M index / B: N index / C: N index
  const int khalf = lane >> 4;   // which K-half this lane holds
  const int rowbase = (blockIdx.x * WAVES_PER_BLOCK + wid) * ROWS_PER_WAVE;

  __shared__ char bstage[2][CHUNK_B];   // padded [96][272B] bf16 chunks
  __shared__ float red[WAVES_PER_BLOCK];

  const float* frow = features + (size_t)(rowbase + row16) * FEAT;

  // Per-thread share of one chunk DMA: 96 rows * 16 units(16B) = 1536 units.
  // unit u: row = u>>4, 16B-slot-in-row = u&15.  256 threads -> 6 units each.
  auto issue_chunk = [&](int kb0, int buf) {
#pragma unroll
    for (int i = 0; i < 6; ++i) {
      const int u   = tid + 256 * i;
      const int row = u >> 4;
      const int uir = u & 15;
      const char* g = (const char*)(ctr_bf + (size_t)row * FEAT + kb0) + uir * 16;
      const unsigned l = (unsigned)(size_t)(&bstage[buf][row * LDS_ROW_B + uir * 16]);
      async_load_b128(g, l);
    }
  };

  v8f acc0 = {}, acc1 = {}, acc2 = {}, acc3 = {}, acc4 = {}, acc5 = {};
  float fsq_part = 0.f;   // f32 partial of ||f_row||^2 (this lane's K subset)

  issue_chunk(0, 0);
  async_wait0();
  __syncthreads();

  for (int kc = 0; kc < N_CHUNKS; ++kc) {
    if (kc + 1 < N_CHUNKS) issue_chunk((kc + 1) * K_CHUNK, (kc + 1) & 1);
    const char* chunk = bstage[kc & 1];

#pragma unroll
    for (int ks = 0; ks < K_CHUNK / 32; ++ks) {
      const int kb = kc * K_CHUNK + ks * 32;

      // ---- A fragment: 16x32 bf16, ISA layout:
      //   VGPR0-3 : K = khalf*8 + {0..7}, VGPR4-7 : K = 16 + khalf*8 + {0..7}
      const float4 c1a = *(const float4*)(frow + kb + khalf * 8);
      const float4 c1b = *(const float4*)(frow + kb + khalf * 8 + 4);
      const float4 c2a = *(const float4*)(frow + kb + 16 + khalf * 8);
      const float4 c2b = *(const float4*)(frow + kb + 16 + khalf * 8 + 4);

      fsq_part += c1a.x * c1a.x + c1a.y * c1a.y + c1a.z * c1a.z + c1a.w * c1a.w
                + c1b.x * c1b.x + c1b.y * c1b.y + c1b.z * c1b.z + c1b.w * c1b.w
                + c2a.x * c2a.x + c2a.y * c2a.y + c2a.z * c2a.z + c2a.w * c2a.w
                + c2b.x * c2b.x + c2b.y * c2b.y + c2b.z * c2b.z + c2b.w * c2b.w;

      v16bf a;
      a[0]  = (__bf16)c1a.x; a[1]  = (__bf16)c1a.y; a[2]  = (__bf16)c1a.z; a[3]  = (__bf16)c1a.w;
      a[4]  = (__bf16)c1b.x; a[5]  = (__bf16)c1b.y; a[6]  = (__bf16)c1b.z; a[7]  = (__bf16)c1b.w;
      a[8]  = (__bf16)c2a.x; a[9]  = (__bf16)c2a.y; a[10] = (__bf16)c2a.z; a[11] = (__bf16)c2a.w;
      a[12] = (__bf16)c2b.x; a[13] = (__bf16)c2b.y; a[14] = (__bf16)c2b.z; a[15] = (__bf16)c2b.w;

      // ---- B fragments from LDS: lane = column (center row ct*16 + row16),
      //      K = ks*32 + khalf*8 + {0..7} and ks*32 + 16 + khalf*8 + {0..7} -> two 16B ds loads.
#define B_FRAG(ct)                                                                     \
      ({                                                                               \
        const char* p = chunk + (size_t)((ct) * 16 + row16) * LDS_ROW_B                \
                               + (size_t)(ks * 32 + khalf * 8) * 2;                    \
        v8bf lo = *(const v8bf*)p;                                                     \
        v8bf hi = *(const v8bf*)(p + 32);                                              \
        __builtin_shufflevector(lo, hi, 0, 1, 2, 3, 4, 5, 6, 7,                        \
                                         8, 9, 10, 11, 12, 13, 14, 15);               \
      })

      v16bf b0 = B_FRAG(0);
      acc0 = __builtin_amdgcn_wmma_f32_16x16x32_bf16(false, a, false, b0, (short)0, acc0, false, false);
      v16bf b1 = B_FRAG(1);
      acc1 = __builtin_amdgcn_wmma_f32_16x16x32_bf16(false, a, false, b1, (short)0, acc1, false, false);
      v16bf b2 = B_FRAG(2);
      acc2 = __builtin_amdgcn_wmma_f32_16x16x32_bf16(false, a, false, b2, (short)0, acc2, false, false);
      v16bf b3 = B_FRAG(3);
      acc3 = __builtin_amdgcn_wmma_f32_16x16x32_bf16(false, a, false, b3, (short)0, acc3, false, false);
      v16bf b4 = B_FRAG(4);
      acc4 = __builtin_amdgcn_wmma_f32_16x16x32_bf16(false, a, false, b4, (short)0, acc4, false, false);
      v16bf b5 = B_FRAG(5);
      acc5 = __builtin_amdgcn_wmma_f32_16x16x32_bf16(false, a, false, b5, (short)0, acc5, false, false);
#undef B_FRAG
    }

    async_wait0();       // chunk kc+1 DMA complete (overlapped with compute above)
    __syncthreads();     // visible to all waves; all reads of buffer kc done
  }

  // ---- Epilogue: pick only the labeled column per row (mask), clamp, reduce.
  // C-tile layout: lane l, reg r  <->  M = r + 8*(l>>4), N = l & 15.
  const int lbl = labels[rowbase + row16];   // lanes 0..15 hold rows 0..15
  float lsum = 0.f;
#pragma unroll
  for (int r = 0; r < 8; ++r) {
    const int m = r + 8 * khalf;
    const float fsqm = __shfl(fsq_part, m, 32) + __shfl(fsq_part, m + 16, 32);
    const int lblm = __shfl(lbl, m, 32);
    if ((lblm & 15) == row16) {              // this lane owns column lblm
      const int ct = lblm >> 4;
      float dot = acc0[r];
      if (ct == 1) dot = acc1[r];
      if (ct == 2) dot = acc2[r];
      if (ct == 3) dot = acc3[r];
      if (ct == 4) dot = acc4[r];
      if (ct == 5) dot = acc5[r];
      float val = fsqm + csq[lblm] - 2.0f * dot;
      val = fminf(fmaxf(val, 1e-12f), 1e12f);
      lsum += val;
    }
  }
#pragma unroll
  for (int off = 16; off > 0; off >>= 1) lsum += __shfl_xor(lsum, off, 32);

  if (lane == 0) red[wid] = lsum;
  __syncthreads();
  if (tid == 0) {
    float tot = 0.f;
#pragma unroll
    for (int w = 0; w < WAVES_PER_BLOCK; ++w) tot += red[w];
    partials[blockIdx.x] = tot;
  }
}

// ---------------------------------------------------------------------------
// Kernel 3: fixed-order tree reduction of 512 block partials -> loss scalar
// ---------------------------------------------------------------------------
__global__ __launch_bounds__(512) void centerloss_finalize(
    const float* __restrict__ partials, float* __restrict__ out) {
  __shared__ float s[512];
  const int t = threadIdx.x;
  s[t] = partials[t];
  __syncthreads();
#pragma unroll
  for (int off = 256; off > 0; off >>= 1) {
    if (t < off) s[t] += s[t + off];
    __syncthreads();
  }
  if (t == 0) out[0] = s[0] * (1.0f / (float)BATCH);
}

// ---------------------------------------------------------------------------
extern "C" void kernel_launch(void* const* d_in, const int* in_sizes, int n_in,
                              void* d_out, int out_size, void* d_ws, size_t ws_size,
                              hipStream_t stream) {
  (void)in_sizes; (void)n_in; (void)out_size; (void)ws_size;
  const float* features = (const float*)d_in[0];
  const float* centers  = (const float*)d_in[1];
  const int*   labels   = (const int*)d_in[2];
  float* out = (float*)d_out;

  char* ws = (char*)d_ws;
  __bf16* ctr_bf   = (__bf16*)ws;                                   // 96*1024*2 B
  float*  csq      = (float*)(ws + (size_t)NCLS * FEAT * 2);        // 96*4 B
  float*  partials = (float*)(ws + (size_t)NCLS * FEAT * 2 + 512);  // 512*4 B

  centerloss_prep<<<NCLS, 256, 0, stream>>>(centers, ctr_bf, csq);
  centerloss_main<<<NBLOCKS, 256, 0, stream>>>(features, labels, ctr_bf, csq, partials);
  centerloss_finalize<<<1, 512, 0, stream>>>(partials, out);
}
